// HyLiFormerBlock_21174188769558
// MI455X (gfx1250) — matrix operand order repaired and checked by reading, not compiled, not run on hardware
//
#include <hip/hip_runtime.h>
#include <hip/hip_bf16.h>

// ---------------------------------------------------------------------------
// Types for CDNA5 WMMA / TDM (gfx1250, wave32)
// ---------------------------------------------------------------------------
typedef __attribute__((ext_vector_type(16))) __bf16 bf16x16;
typedef __attribute__((ext_vector_type(8)))  float  f32x8;
typedef __attribute__((ext_vector_type(4))) unsigned int u32x4;
typedef __attribute__((ext_vector_type(8))) int         i32x8;
typedef __attribute__((ext_vector_type(4))) int         i32x4;

struct __attribute__((aligned(16))) Q16 { unsigned int w[4]; };
union ABu { bf16x16 v; Q16 q[2]; unsigned short u[16]; };

__device__ __forceinline__ unsigned short f2bfu(float f) {
  unsigned u = __float_as_uint(f);
  unsigned r = u + 0x7FFFu + ((u >> 16) & 1u);   // round-to-nearest-even
  return (unsigned short)(r >> 16);
}

// Problem constants
#define BB   32
#define CC   256
#define TT   64
#define VV   50
#define NTOK (BB * TT * VV)   // 102400

// ---------------------------------------------------------------------------
// 1) x (B,C,T,V) -> xt (NTOK, C) token-major  (coalesced 32x32 LDS transpose)
// ---------------------------------------------------------------------------
__global__ __launch_bounds__(256) void transpose_kernel(const float* __restrict__ x,
                                                        float* __restrict__ xt) {
  __shared__ float tile[32][33];
  const int b  = blockIdx.z;
  const int c0 = blockIdx.y * 32;
  const int s0 = blockIdx.x * 32;            // spatial index t*V+v, 3200 = 100*32
  const int tx = threadIdx.x, ty = threadIdx.y;
#pragma unroll
  for (int k = 0; k < 4; ++k)
    tile[ty + k * 8][tx] = x[((size_t)b * CC + c0 + ty + k * 8) * (TT * VV) + s0 + tx];
  __syncthreads();
#pragma unroll
  for (int k = 0; k < 4; ++k)
    xt[((size_t)b * (TT * VV) + s0 + ty + k * 8) * CC + c0 + tx] = tile[tx][ty + k * 8];
}

// ---------------------------------------------------------------------------
// 2) Pack fp32 weight W[K][N] into WMMA-B bf16 fragment layout:
//    P[((nt*Kt + kt)*32 + lane)*16 + i] = W[kt*32 + i + (lane>>4)*16][nt*16 + (lane&15)]
// ---------------------------------------------------------------------------
__global__ __launch_bounds__(256) void pack_w_kernel(const float* __restrict__ W,
                                                     unsigned short* __restrict__ P,
                                                     int K, int Nn) {
  int idx = blockIdx.x * 256 + threadIdx.x;
  if (idx >= K * Nn) return;
  int i    = idx & 15;
  int lane = (idx >> 4) & 31;
  int tile = idx >> 9;
  int Kt   = K >> 5;
  int kt   = tile % Kt;
  int nt   = tile / Kt;
  int k    = kt * 32 + i + (lane >> 4) * 16;
  int n    = nt * 16 + (lane & 15);
  P[idx] = f2bfu(W[(size_t)k * Nn + n]);
}

// ---------------------------------------------------------------------------
// 3) LayerNorm (wave per token, 8 channels per lane) -> bf16 row
// ---------------------------------------------------------------------------
__global__ __launch_bounds__(256) void ln_kernel(const float* __restrict__ X,
                                                 const float* __restrict__ w,
                                                 const float* __restrict__ bb,
                                                 unsigned short* __restrict__ Y) {
  const int tok  = (blockIdx.x * 256 + threadIdx.x) >> 5;
  const int lane = threadIdx.x & 31;
  const float* xr = X + (size_t)tok * CC;
  float v[8]; float s = 0.f;
#pragma unroll
  for (int i = 0; i < 8; ++i) { v[i] = xr[lane * 8 + i]; s += v[i]; }
#pragma unroll
  for (int o = 16; o > 0; o >>= 1) s += __shfl_xor(s, o, 32);
  const float mean = s * (1.f / 256.f);
  float q = 0.f;
#pragma unroll
  for (int i = 0; i < 8; ++i) { float d = v[i] - mean; q += d * d; }
#pragma unroll
  for (int o = 16; o > 0; o >>= 1) q += __shfl_xor(q, o, 32);
  const float inv = rsqrtf(q * (1.f / 256.f) + 1e-5f);
  unsigned short* yr = Y + (size_t)tok * CC;
#pragma unroll
  for (int i = 0; i < 8; ++i) {
    int c = lane * 8 + i;
    yr[c] = f2bfu((v[i] - mean) * inv * w[c] + bb[c]);
  }
}

// ---------------------------------------------------------------------------
// 4) WMMA bf16 GEMM with TDM-staged B tiles.
//    Block = 256 thr = 8 waves; block tile 256(M) x 64(N); wave tile 32 x 64.
//    B fragments (4 n-tiles x 512 bf16 = 4KB per k-step) are DMA'd into LDS by
//    the Tensor Data Mover (wave 0 issues tensor_load_to_lds; D# describes a
//    2-D tile: 4 rows of 512 elements, row stride = Kt*512 elements), double
//    buffered, synchronized with s_wait_tensorcnt + barrier. A fragments come
//    straight from global (pipelined b128 loads).
//    mode 0: f          = acc + bias                     (fp32, Ntot=512)
//    mode 1: res1       = acc + bias + xt (skip)         (fp32, Ntot=256)
//    mode 2: hmid       = gelu(acc + bias)               (bf16, Ntot=1024)
//    mode 3: d_out[B,C,T,V] = acc + bias + res1          (fp32, Ntot=256)
// ---------------------------------------------------------------------------
__device__ __forceinline__ bf16x16 load_a_frag(const unsigned short* __restrict__ A,
                                               int lda, int row, int k0, int hi) {
  // 16-bit A 16x32 (contiguous, matches packed B): lanes 0-15 K{0-15}, lanes 16-31 K{16-31}
  const unsigned short* p = A + (size_t)row * lda + k0 + hi * 16;
  ABu r;
  r.q[0] = *(const Q16*)p;          // u[0..7]  -> K = k0+hi*16 .. +7
  r.q[1] = *(const Q16*)(p + 8);    // u[8..15] -> K = k0+hi*16+8 .. +7
  return r.v;
}

// Issue one TDM load: 4 rows x 512 bf16, row stride Kt*512 elems -> lds_off.
// D# layout per CDNA5 ISA.
__device__ __forceinline__ void tdm_load_b(const unsigned short* gsrc,
                                           unsigned lds_off, int Kt) {
  unsigned long long ga = (unsigned long long)gsrc;
  unsigned long long st = (unsigned long long)Kt * 512ull;   // elements
  u32x4 g0;
  g0[0] = 1u;                                    // count=1, user descriptor
  g0[1] = lds_off;                               // lds_addr (bytes)
  g0[2] = (unsigned)ga;                          // global_addr[31:0]
  g0[3] = (unsigned)(ga >> 32) | (2u << 30);     // global_addr[56:32] | type=2
  i32x8 g1;
  g1[0] = (int)(1u << 16);                       // wg_mask=0, data_size=1 (2B)
  g1[1] = (int)(512u << 16);                     // tensor_dim0[15:0] in [31:16]
  g1[2] = (int)((512u >> 16) | (4u << 16));      // tensor_dim0 hi | tensor_dim1 lo
  g1[3] = (int)(512u << 16);                     // tensor_dim1 hi=0 | tile_dim0=512
  g1[4] = (int)4u;                               // tile_dim1=4, tile_dim2=0
  g1[5] = (int)(unsigned)st;                     // dim0 stride [31:0]
  g1[6] = (int)(unsigned)((st >> 32) & 0xFFFFu); // dim0 stride [47:32]
  g1[7] = 0;
  i32x4 z4 = {0, 0, 0, 0};
#if __clang_major__ >= 23
  i32x8 z8 = {0, 0, 0, 0, 0, 0, 0, 0};
  __builtin_amdgcn_tensor_load_to_lds(g0, g1, z4, z4, z8, 0);
#else
  __builtin_amdgcn_tensor_load_to_lds(g0, g1, z4, z4, 0);
#endif
}

__global__ __launch_bounds__(256) void gemm_kernel(const unsigned short* __restrict__ A,
                                                   const unsigned short* __restrict__ Wp,
                                                   const float* __restrict__ bias,
                                                   int K, int Ntot, int mode,
                                                   float* __restrict__ out_f32,
                                                   unsigned short* __restrict__ out_bf16,
                                                   const float* __restrict__ aux) {
  __shared__ unsigned short Bsh[2][2048];        // double-buffered 4KB B stage
  const int tid  = threadIdx.x;
  const int wave = tid >> 5;
  const int lane = tid & 31;
  const int hi   = lane >> 4;
  const int lm   = lane & 15;
  const int rowbase = blockIdx.x * 256 + wave * 32;
  const int nt0     = blockIdx.y * 4;
  const int Kt      = K >> 5;

  const unsigned short* bbase = Wp + (size_t)nt0 * Kt * 512;   // 4 row-tiles, stride Kt*512
  const unsigned lds0 = (unsigned)(size_t)&Bsh[0][0];          // LDS byte offset
  const unsigned lds1 = (unsigned)(size_t)&Bsh[1][0];

  if (wave == 0) {
    tdm_load_b(bbase, lds0, Kt);                  // k-step 0 -> stage 0
    __builtin_amdgcn_s_wait_tensorcnt(0);
  }
  __syncthreads();

  f32x8 acc[2][4] = {};
  for (int kt = 0; kt < Kt; ++kt) {
    const int cur = kt & 1;
    if (wave == 0 && kt + 1 < Kt)                 // prefetch next stage via TDM
      tdm_load_b(bbase + (size_t)(kt + 1) * 512, cur ? lds0 : lds1, Kt);

    bf16x16 a0 = load_a_frag(A, K, rowbase + lm,      kt * 32, hi);
    bf16x16 a1 = load_a_frag(A, K, rowbase + 16 + lm, kt * 32, hi);
#pragma unroll
    for (int j = 0; j < 4; ++j) {
      ABu bf;                                      // ds_load_b128 x2 from stage
      bf.q[0] = *(const Q16*)&Bsh[cur][j * 512 + lane * 16];
      bf.q[1] = *(const Q16*)&Bsh[cur][j * 512 + lane * 16 + 8];
      acc[0][j] = __builtin_amdgcn_wmma_f32_16x16x32_bf16(
          false, a0, false, bf.v, (short)0, acc[0][j], false, false);
      acc[1][j] = __builtin_amdgcn_wmma_f32_16x16x32_bf16(
          false, a1, false, bf.v, (short)0, acc[1][j], false, false);
    }
    if (kt + 1 < Kt) {
      if (wave == 0) __builtin_amdgcn_s_wait_tensorcnt(0);
      __syncthreads();                             // stage ready + readers done
    }
  }

#pragma unroll
  for (int g = 0; g < 2; ++g)
#pragma unroll
    for (int j = 0; j < 4; ++j) {
      const int col = (nt0 + j) * 16 + lm;
      const float bv = bias[col];
#pragma unroll
      for (int r = 0; r < 8; ++r) {
        const int row = rowbase + g * 16 + r + hi * 8;   // D: m = r + hi*8
        float val = acc[g][j][r] + bv;
        if (mode == 0) {
          out_f32[(size_t)row * Ntot + col] = val;
        } else if (mode == 1) {
          out_f32[(size_t)row * CC + col] = val + aux[(size_t)row * CC + col];
        } else if (mode == 2) {
          float ge = 0.5f * val * (1.f + erff(val * 0.70710678118654752f));
          out_bf16[(size_t)row * Ntot + col] = f2bfu(ge);
        } else {
          val += aux[(size_t)row * CC + col];
          int b = row / (TT * VV);
          int r2 = row - b * (TT * VV);
          int t = r2 / VV, v = r2 - t * VV;
          out_f32[(((size_t)b * CC + col) * TT + t) * VV + v] = val;
        }
      }
    }
}

// ---------------------------------------------------------------------------
// 5) Graph conv: cat[:, 0:64] -- y_g[b,c,t,v] = sum_u f[tok(b,t,u)][c]*gw[g,v,u]
// ---------------------------------------------------------------------------
__global__ __launch_bounds__(256) void gconv_kernel(const float* __restrict__ f,
                                                    const float* __restrict__ gw,
                                                    unsigned short* __restrict__ cat) {
  __shared__ float fg[VV * 64];
  const int bt  = blockIdx.x;               // b*T + t
  const int tid = threadIdx.x;
  for (int i = tid; i < VV * 64; i += 256) {
    int u = i >> 6, c = i & 63;
    fg[u * 64 + c] = f[((size_t)bt * VV + u) * 512 + c];
  }
  __syncthreads();
  for (int i = tid; i < VV * 64; i += 256) {
    int c = i & 63, v = i >> 6, g = c >> 3;
    const float* wr = gw + ((size_t)g * VV + v) * VV;
    float s = 0.f;
    for (int u = 0; u < VV; ++u) s += fg[u * 64 + c] * wr[u];
    cat[((size_t)bt * VV + v) * CC + c] = f2bfu(s);
  }
}

// ---------------------------------------------------------------------------
// 6) Temporal grouped conv 7x1, pad 3: cat[:, 64:128]
// ---------------------------------------------------------------------------
__global__ __launch_bounds__(256) void tconv_kernel(const float* __restrict__ f,
                                                    const float* __restrict__ tw,
                                                    const float* __restrict__ tb,
                                                    unsigned short* __restrict__ cat) {
  int idx = blockIdx.x * 256 + threadIdx.x;  // BB*TT*VV*64 = 6,553,600 threads
  int co = idx & 63;
  int r  = idx >> 6;
  int v  = r % VV;
  int t  = (r / VV) % TT;
  int b  = r / (TT * VV);
  int g  = co >> 3;
  float s = tb[co];
#pragma unroll
  for (int ci = 0; ci < 8; ++ci) {
    const float* wp = tw + ((size_t)co * 8 + ci) * 7;
#pragma unroll
    for (int kt = 0; kt < 7; ++kt) {
      int t2 = t + kt - 3;
      if (t2 >= 0 && t2 < TT)
        s += f[(((size_t)b * TT + t2) * VV + v) * 512 + 64 + g * 8 + ci] * wp[kt];
    }
  }
  cat[(((size_t)b * TT + t) * VV + v) * CC + 64 + co] = f2bfu(s);
}

// ---------------------------------------------------------------------------
// 7) Hyperbolic windowed linear attention: one block per (window, chunk)
//    window = 80 tokens x 96 ch; heads h=4, d=8.  Writes cat[:, 128+chunk*32 ..]
// ---------------------------------------------------------------------------
__global__ __launch_bounds__(256) void attn_kernel(const float* __restrict__ f,
                                                   unsigned short* __restrict__ cat) {
  const int chunk = blockIdx.y;             // mode = chunk + 1
  const int win   = blockIdx.x;             // 0..1279 = B * (T/8) * (V/10)
  const int b     = win / 40;
  const int rem   = win % 40;
  const int ti    = rem / 5;
  const int vi    = rem % 5;
  const int tid   = threadIdx.x;

  __shared__ float fw[80 * 97];             // padded stride vs 64 banks
  __shared__ float kvm[4][8][8];

  auto tok_of = [&](int n) {
    int a = n / 10, b2 = n % 10;
    int t = (chunk < 2) ? (ti * 8 + a) : (a * 8 + ti);
    int v = ((chunk & 1) == 0) ? (vi * 10 + b2) : (b2 * 5 + vi);
    return (b * TT + t) * VV + v;
  };

  for (int i = tid; i < 80 * 96; i += 256) {
    int n = i / 96, c = i % 96;
    fw[n * 97 + c] = f[(size_t)tok_of(n) * 512 + 128 + chunk * 96 + c];
  }
  __syncthreads();

  if (tid < 80) {                           // euclidean -> poincare
    float ss = 0.f;
    for (int c = 0; c < 96; ++c) { float x = fw[tid * 97 + c]; ss += x * x; }
    float nn = fmaxf(sqrtf(ss), 1e-10f);
    float sc = tanhf(nn) / nn;
    for (int c = 0; c < 96; ++c) fw[tid * 97 + c] *= sc;
  }
  __syncthreads();

  {                                         // kv[d][e] = sum_n k[n,d] v[n,e]
    int head = tid >> 6, d = (tid >> 3) & 7, e = tid & 7;
    float s = 0.f;
    for (int n = 0; n < 80; ++n)
      s += fw[n * 97 + 32 + head * 8 + d] * fw[n * 97 + 64 + head * 8 + e];
    kvm[head][d][e] = s;
  }
  __syncthreads();
  if (tid < 32) {                           // softmax over e
    int head = tid >> 3, d = tid & 7;
    float mx = -1e30f;
    for (int e = 0; e < 8; ++e) mx = fmaxf(mx, kvm[head][d][e]);
    float ex[8], sum = 0.f;
    for (int e = 0; e < 8; ++e) { ex[e] = __expf(kvm[head][d][e] - mx); sum += ex[e]; }
    float inv = 1.f / sum;
    for (int e = 0; e < 8; ++e) kvm[head][d][e] = ex[e] * inv;
  }
  __syncthreads();

  if (tid < 80) {                           // attn = q @ kv; poincare -> euclid
    float o[32]; float ss = 0.f;
#pragma unroll
    for (int head = 0; head < 4; ++head)
#pragma unroll
      for (int m = 0; m < 8; ++m) {
        float s = 0.f;
#pragma unroll
        for (int i = 0; i < 8; ++i)
          s += 0.5f * fw[tid * 97 + head * 8 + i] * kvm[head][m][i];
        o[head * 8 + m] = s; ss += s * s;
      }
    float nn = fminf(fmaxf(sqrtf(ss), 1e-10f), 1.f - 1e-5f);
    float sc = atanhf(-nn) / nn;            // reference: arctanh(n * (-1/CURV))
    size_t base = (size_t)tok_of(tid) * CC + 128 + chunk * 32;
    for (int c2 = 0; c2 < 32; ++c2) cat[base + c2] = f2bfu(o[c2] * sc);
  }
}

// ---------------------------------------------------------------------------
// Host launcher
// ---------------------------------------------------------------------------
extern "C" void kernel_launch(void* const* d_in, const int* in_sizes, int n_in,
                              void* d_out, int out_size, void* d_ws, size_t ws_size,
                              hipStream_t stream) {
  const float* x       = (const float*)d_in[0];
  const float* norm1_w = (const float*)d_in[1];
  const float* norm1_b = (const float*)d_in[2];
  const float* map_w   = (const float*)d_in[3];
  const float* map_b   = (const float*)d_in[4];
  const float* gconv   = (const float*)d_in[5];
  const float* tconv_w = (const float*)d_in[6];
  const float* tconv_b = (const float*)d_in[7];
  const float* proj_w  = (const float*)d_in[8];
  const float* proj_b  = (const float*)d_in[9];
  const float* norm2_w = (const float*)d_in[10];
  const float* norm2_b = (const float*)d_in[11];
  const float* mlp_w1  = (const float*)d_in[12];
  const float* mlp_b1  = (const float*)d_in[13];
  const float* mlp_w2  = (const float*)d_in[14];
  const float* mlp_b2  = (const float*)d_in[15];

  char* ws = (char*)d_ws;
  size_t off = 0;
  float*          xt   = (float*)(ws + off);           off += (size_t)NTOK * CC * 4;    // 104.8 MB
  unsigned short* xn   = (unsigned short*)(ws + off);  off += (size_t)NTOK * CC * 2;    //  52.4 MB
  float*          fbuf = (float*)(ws + off);           off += (size_t)NTOK * 512 * 4;   // 209.7 MB
  unsigned short* cat  = (unsigned short*)(ws + off);  off += (size_t)NTOK * CC * 2;    //  52.4 MB
  float*          res1 = (float*)(ws + off);           off += (size_t)NTOK * CC * 4;    // 104.8 MB
  unsigned short* ln2b = (unsigned short*)(ws + off);  off += (size_t)NTOK * CC * 2;    //  52.4 MB
  unsigned short* hmid = (unsigned short*)(ws + off);  off += (size_t)NTOK * 1024 * 2;  // 209.7 MB
  unsigned short* mapP = (unsigned short*)(ws + off);  off += (size_t)256 * 512 * 2;
  unsigned short* prjP = (unsigned short*)(ws + off);  off += (size_t)256 * 256 * 2;
  unsigned short* m1P  = (unsigned short*)(ws + off);  off += (size_t)256 * 1024 * 2;
  unsigned short* m2P  = (unsigned short*)(ws + off);  off += (size_t)1024 * 256 * 2;

  // 1) token-major transpose of x (also used as residual skip)
  transpose_kernel<<<dim3(100, 8, BB), dim3(32, 8), 0, stream>>>(x, xt);

  // 2) weight packing into WMMA-B fragment layout
  pack_w_kernel<<<(256 * 512 + 255) / 256, 256, 0, stream>>>(map_w, mapP, 256, 512);
  pack_w_kernel<<<(256 * 256 + 255) / 256, 256, 0, stream>>>(proj_w, prjP, 256, 256);
  pack_w_kernel<<<(256 * 1024 + 255) / 256, 256, 0, stream>>>(mlp_w1, m1P, 256, 1024);
  pack_w_kernel<<<(1024 * 256 + 255) / 256, 256, 0, stream>>>(mlp_w2, m2P, 1024, 256);

  // 3) LN1 -> bf16 tokens
  ln_kernel<<<NTOK / 8, 256, 0, stream>>>(xt, norm1_w, norm1_b, xn);

  // 4) map GEMM: f = LN1(x) @ map_w + map_b   (N x 512, fp32)
  gemm_kernel<<<dim3(NTOK / 256, 512 / 64), 256, 0, stream>>>(
      xn, mapP, map_b, 256, 512, 0, fbuf, nullptr, nullptr);

  // 5-7) branch ops -> cat (bf16, N x 256)
  gconv_kernel<<<BB * TT, 256, 0, stream>>>(fbuf, gconv, cat);
  tconv_kernel<<<(BB * TT * VV * 64) / 256, 256, 0, stream>>>(fbuf, tconv_w, tconv_b, cat);
  attn_kernel<<<dim3(BB * (TT / 8) * (VV / 10), 4), 256, 0, stream>>>(fbuf, cat);

  // 8) proj GEMM + skip: res1 = cat @ proj_w + proj_b + xt
  gemm_kernel<<<dim3(NTOK / 256, 256 / 64), 256, 0, stream>>>(
      cat, prjP, proj_b, 256, 256, 1, res1, nullptr, xt);

  // 9) LN2 -> bf16
  ln_kernel<<<NTOK / 8, 256, 0, stream>>>(res1, norm2_w, norm2_b, ln2b);

  // 10) MLP up + exact GELU -> bf16 hmid (N x 1024)
  gemm_kernel<<<dim3(NTOK / 256, 1024 / 64), 256, 0, stream>>>(
      ln2b, m1P, mlp_b1, 256, 1024, 2, nullptr, hmid, nullptr);

  // 11) MLP down + residual, written transposed to d_out (B,C,T,V)
  gemm_kernel<<<dim3(NTOK / 256, 256 / 64), 256, 0, stream>>>(
      hmid, m2P, mlp_b2, 1024, 256, 3, (float*)d_out, nullptr, res1);

  (void)in_sizes; (void)n_in; (void)out_size; (void)ws_size;
}
